// BaGuaLLM_26079041421884
// MI455X (gfx1250) — matrix operand order repaired and checked by reading, not compiled, not run on hardware
//
#include <hip/hip_runtime.h>
#include <hip/hip_fp16.h>

typedef _Float16 h16;
typedef __attribute__((ext_vector_type(16))) _Float16 v16h;
typedef __attribute__((ext_vector_type(8)))  float    v8f;

#define DIMD 768
#define NHD 8
#define HDD 96
#define LLAY 12
#define RD 2048      /* B*S */
#define VOCABD 32000

__constant__ float c_scene_w[23] = {0.95f,0.95f,0.9f,0.85f,0.8f,0.95f,0.9f,0.05f,0.05f,0.1f,0.1f,0.05f,0.05f,0.05f,0.1f,0.05f,0.4f,0.7f,0.35f,0.45f,0.6f,0.5f,0.65f};

__device__ __forceinline__ float geluf(float x){ return 0.5f*x*(1.0f+erff(x*0.70710678118654752f)); }
__device__ __forceinline__ float sigmf(float x){ return 1.0f/(1.0f+expf(-x)); }

// ---------------------------------------------------------------------------
// Generic GEMM: C[M,N](f32) = act( (A[M,K] f32 @ B[K,N] f16)*colscale + bias + res )
// BM=128, BN=128, BK=32, 256 threads = 8 waves; wave w owns rows w*16..w*16+15
// and all 128 cols (8 WMMAs / K-step, A fragment reused 8x).
// Double-buffered LDS staging: next tile's global loads issue before the
// barrier and retire under the WMMAs of the current step.
// ACT: 0 none, 1 gelu, 2 sigmoid, 3 tanh
// ---------------------------------------------------------------------------
template<int ACT>
__launch_bounds__(256)
__global__ void gemm_kernel(const float* __restrict__ A, const h16* __restrict__ Bw,
                            float* __restrict__ C, int M, int N, int K,
                            const float* __restrict__ bias,
                            const float* __restrict__ res,
                            const float* __restrict__ cscale)
{
    __shared__ __align__(16) h16 sA[2][128*40];
    __shared__ __align__(16) h16 sB[2][128*40];
    const int tid  = threadIdx.x;
    const int lane = tid & 31;
    const int wv   = tid >> 5;
    const int mB   = blockIdx.y * 128;
    const int nB   = blockIdx.x * 128;

    v8f acc[8];
    #pragma unroll
    for (int t = 0; t < 8; t++) acc[t] = (v8f){0.f,0.f,0.f,0.f,0.f,0.f,0.f,0.f};

    union U8h { uint4 u; h16 h[8]; };
    union V16 { v16h v; uint4 u[2]; };

    // staging registers (tile in flight)
    float4 ra[4];
    uint4  rb[2];

    // A: thread covers row tid/2, half-row (tid&1)*16 .. +15   (128x32 f32)
    const int arow = tid >> 1;
    const int acb  = (tid & 1) * 16;
    // B: thread covers k-row tid/8, cols (tid&7)*16 .. +15      (32x128 f16)
    const int brow = tid >> 3;
    const int bcb  = (tid & 7) * 16;
    const int bgc  = nB + bcb;

    auto load_global = [&](int k0){
        const float* ga = A + (size_t)(mB + arow) * K + k0 + acb;
        #pragma unroll
        for (int i = 0; i < 4; i++) ra[i] = *(const float4*)(ga + i*4);
        const h16* gb = Bw + (size_t)(k0 + brow) * N + bgc;
        if (bgc + 16 <= N) {
            rb[0] = *(const uint4*)gb;
            rb[1] = *(const uint4*)(gb + 8);
            __builtin_prefetch(gb + (size_t)32 * N, 0, 1);   // next K-step of B
        } else {
            U8h t0, t1;
            #pragma unroll
            for (int i = 0; i < 8; i++) {
                t0.h[i] = (bgc + i     < N) ? gb[i]     : (h16)0.0f;
                t1.h[i] = (bgc + 8 + i < N) ? gb[8 + i] : (h16)0.0f;
            }
            rb[0] = t0.u; rb[1] = t1.u;
        }
    };
    auto store_lds = [&](int buf){
        h16* la = sA[buf] + arow*40 + acb;
        #pragma unroll
        for (int i = 0; i < 4; i++) {
            la[i*4+0] = (h16)ra[i].x; la[i*4+1] = (h16)ra[i].y;
            la[i*4+2] = (h16)ra[i].z; la[i*4+3] = (h16)ra[i].w;
        }
        U8h u0, u1; u0.u = rb[0]; u1.u = rb[1];
        h16* lb = sB[buf];
        #pragma unroll
        for (int i = 0; i < 8; i++) {
            lb[(bcb + i    )*40 + brow] = u0.h[i];
            lb[(bcb + 8 + i)*40 + brow] = u1.h[i];
        }
    };

    // prologue: stage tile 0
    load_global(0);
    store_lds(0);

    int cur = 0;
    for (int k0 = 0; k0 < K; k0 += 32) {
        const bool has_next = (k0 + 32 < K);
        if (has_next) load_global(k0 + 32);      // overlaps with compute below
        __syncthreads();                         // lds[cur] ready; prev readers of lds[cur^1] done

        // ---- A fragment (16x32, ISA layout) ----
        V16 af;
        {
            const int mrow = wv*16 + (lane & 15);
            const int koff = (lane >> 4) * 8;
            const h16* ap = sA[cur] + mrow*40 + koff;
            af.u[0] = *(const uint4*)ap;
            af.u[1] = *(const uint4*)(ap + 16);
        }
        // ---- 8 x (B fragment + WMMA), A reused ----
        #pragma unroll
        for (int t = 0; t < 8; t++) {
            V16 bf;
            const int col  = t*16 + (lane & 15);
            const int koff = (lane >> 4) * 8;
            const h16* bp = sB[cur] + col*40 + koff;
            bf.u[0] = *(const uint4*)bp;
            bf.u[1] = *(const uint4*)(bp + 16);
            acc[t] = __builtin_amdgcn_wmma_f32_16x16x32_f16(
                false, af.v, false, bf.v, (short)0, acc[t], false, false);
        }
        if (has_next) store_lds(cur ^ 1);        // writes other buffer; safe pre-barrier
        cur ^= 1;
    }

    // ---- epilogue (C layout: lane<16 rows m0+g, lane>=16 rows m0+8+g) ----
    const int rbase = mB + wv*16 + ((lane >> 4) << 3);
    #pragma unroll
    for (int t = 0; t < 8; t++) {
        const int col = nB + t*16 + (lane & 15);
        if (col >= N) continue;
        const float cs = cscale ? cscale[col] : 1.0f;
        const float bb = bias   ? bias[col]   : 0.0f;
        #pragma unroll
        for (int g = 0; g < 8; g++) {
            const int r = rbase + g;
            float v = acc[t][g] * cs + bb;
            if (res) v += res[(size_t)r * N + col];
            if (ACT == 1) v = geluf(v);
            else if (ACT == 2) v = sigmf(v);
            else if (ACT == 3) v = tanhf(v);
            C[(size_t)r * N + col] = v;
        }
    }
}

// ---------------------------------------------------------------------------
// Small kernels
// ---------------------------------------------------------------------------
__global__ void f32_to_f16_kernel(const float* __restrict__ s, h16* __restrict__ d, size_t n){
    size_t i = (size_t)blockIdx.x*blockDim.x + threadIdx.x;
    size_t st = (size_t)gridDim.x*blockDim.x;
    for (; i < n; i += st) d[i] = (h16)s[i];
}

// trig_w [L,NH,DIM,HD] -> per layer f16 [DIM, NH*HD]
__global__ void trig_perm_kernel(const float* __restrict__ src, h16* __restrict__ dst){
    int idx = blockIdx.x*blockDim.x + threadIdx.x;
    if (idx >= LLAY*DIMD*DIMD) return;
    int l = idx / (DIMD*DIMD);
    int rem = idx % (DIMD*DIMD);
    int d = rem / DIMD;
    int c = rem % DIMD;
    int n = c / HDD, h = c % HDD;
    dst[idx] = (h16)src[(((size_t)l*NHD + n)*DIMD + d)*HDD + h];
}

// colscale[l][n*96+h] = cos(res_freq * pi)
__global__ void colscale_kernel(const float* __restrict__ rf, float* __restrict__ cs){
    int idx = blockIdx.x*blockDim.x + threadIdx.x;
    if (idx >= LLAY*DIMD) return;
    int l = idx / DIMD, c = idx % DIMD;
    int n = c / HDD, h = c % HDD;
    cs[idx] = cosf(rf[((size_t)l*NHD + n)*HDD + h] * 3.14159265358979323846f);
}

__global__ void embed_kernel(const int* __restrict__ ids, const float* __restrict__ emb,
                             const float* __restrict__ pos, float* __restrict__ x){
    int idx = blockIdx.x*256 + threadIdx.x;     // exact grid RD*DIMD
    int bs = idx / DIMD, d = idx % DIMD;
    int s = bs & 255;
    int tok = ids[bs];
    x[idx] = emb[(size_t)tok*DIMD + d] + pos[(size_t)s*DIMD + d];
}

__global__ void scene_kernel(const float* __restrict__ x,
                             const float* __restrict__ w1, const float* __restrict__ b1,
                             const float* __restrict__ w2, const float* __restrict__ b2,
                             float* __restrict__ factor){
    __shared__ float t[384];
    __shared__ float logits[23];
    __shared__ float accum;
    int tid = threadIdx.x;
    if (tid == 0) accum = 0.f;
    for (int b = 0; b < 8; b++) {
        const float* xr = x + (size_t)b*256*DIMD;   // CLS token row
        __syncthreads();
        for (int j = tid; j < 384; j += 256) {
            float a = b1[j];
            for (int d = 0; d < DIMD; d++) a += xr[d]*w1[(size_t)d*384 + j];
            t[j] = geluf(a);
        }
        __syncthreads();
        if (tid < 23) {
            float a = b2[tid];
            for (int j = 0; j < 384; j++) a += t[j]*w2[(size_t)j*23 + tid];
            logits[tid] = a;
        }
        __syncthreads();
        if (tid == 0) {
            float mx = -1e30f;
            for (int c = 0; c < 23; c++) mx = fmaxf(mx, logits[c]);
            float se = 0.f;
            for (int c = 0; c < 23; c++) se += expf(logits[c]-mx);
            float dot = 0.f;
            for (int c = 0; c < 23; c++) dot += (expf(logits[c]-mx)/se) * c_scene_w[c];
            accum += dot;
        }
    }
    __syncthreads();
    bool causal = (accum / 8.0f) > 0.5f;
    for (int s = tid; s < 256; s += 256)
        factor[s] = causal ? (float)(s+1) / 256.0f : 1.0f;
}

__global__ void hm_kernel(const float* __restrict__ H, float* __restrict__ hm){
    int idx = blockIdx.x*256 + threadIdx.x;
    if (idx >= 8*DIMD) return;
    int b = idx / DIMD, d = idx % DIMD;
    float s = 0.f;
    const float* p = H + (size_t)b*256*DIMD + d;
    for (int t = 0; t < 256; t++) s += p[(size_t)t*DIMD];
    hm[idx] = s / 256.0f;
}

__global__ void coef_kernel(const float* __restrict__ hm,
                            const float* __restrict__ pol_w, const float* __restrict__ pol_b,
                            const float* __restrict__ iw1, const float* __restrict__ ib1,
                            const float* __restrict__ iw2, const float* __restrict__ ib2,
                            float* __restrict__ coef){
    __shared__ float pv[8*8*32];
    __shared__ float rn[64];
    int tid = threadIdx.x;
    for (int idx = tid; idx < 2048; idx += 256) {
        int b = idx >> 8, n = (idx >> 5) & 7, p = idx & 31;
        float a = pol_b[n*32 + p];
        const float* h = hm + b*DIMD + n*HDD;
        const float* w = pol_w + (size_t)(n*HDD)*32 + p;
        for (int k = 0; k < HDD; k++) a += h[k]*w[(size_t)k*32];
        pv[(b*8+n)*32 + p] = tanhf(a);
    }
    __syncthreads();
    if (tid < 64) {
        float s = 0.f;
        const float* v = pv + tid*32;
        for (int p = 0; p < 32; p++) s += v[p]*v[p];
        rn[tid] = 1.0f / fmaxf(sqrtf(s), 1e-12f);
    }
    __syncthreads();
    for (int idx = tid; idx < 512; idx += 256) {
        int b = idx >> 6, i = (idx >> 3) & 7, j = idx & 7;
        float d = 0.f;
        const float* vi = pv + (b*8+i)*32;
        const float* vj = pv + (b*8+j)*32;
        for (int p = 0; p < 32; p++) d += vi[p]*vj[p];
        d *= rn[b*8+i]*rn[b*8+j];
        float s2 = ib2[0];
        for (int k = 0; k < 16; k++) s2 += geluf(d*iw1[k] + ib1[k]) * iw2[k];
        float sp = (s2 > 20.f) ? s2 : log1pf(expf(s2));
        coef[idx] = (i != j) ? 0.1f / (1.0f + sp) : 0.0f;
    }
}

__global__ void transfer_kernel(const float* __restrict__ H, const float* __restrict__ coef,
                                const float* __restrict__ factor, float* __restrict__ M){
    int idx = blockIdx.x*256 + threadIdx.x;
    if (idx >= RD*DIMD) return;
    int bs = idx / DIMD, d = idx % DIMD;
    int b = bs >> 8, s = bs & 255;
    int i = d / HDD, h = d % HDD;
    float f = factor[s];
    const float* cf = coef + b*64 + i*8;
    const float* hr = H + (size_t)bs*DIMD + h;
    float a = H[idx];
    #pragma unroll
    for (int j = 0; j < 8; j++) a += f * cf[j] * hr[(size_t)j*HDD];
    M[idx] = a;
}

__global__ void ln_kernel(const float* __restrict__ in, float* __restrict__ out,
                          const float* __restrict__ g, const float* __restrict__ b){
    __shared__ float red[256];
    __shared__ float s_mean, s_rstd;
    int row = blockIdx.x, tid = threadIdx.x;
    const float* p = in + (size_t)row*DIMD;
    float s = 0.f, s2 = 0.f;
    for (int d = tid; d < DIMD; d += 256) { float v = p[d]; s += v; s2 += v*v; }
    red[tid] = s; __syncthreads();
    for (int o = 128; o > 0; o >>= 1){ if (tid < o) red[tid] += red[tid+o]; __syncthreads(); }
    if (tid == 0) s_mean = red[0] / (float)DIMD;
    __syncthreads();
    red[tid] = s2; __syncthreads();
    for (int o = 128; o > 0; o >>= 1){ if (tid < o) red[tid] += red[tid+o]; __syncthreads(); }
    if (tid == 0) {
        float var = red[0] / (float)DIMD - s_mean*s_mean;
        s_rstd = rsqrtf(var + 1e-5f);
    }
    __syncthreads();
    float mean = s_mean, rstd = s_rstd;
    for (int d = tid; d < DIMD; d += 256)
        out[(size_t)row*DIMD + d] = (p[d]-mean)*rstd*g[d] + b[d];
}

__global__ void zone_kernel(const float* __restrict__ bagua, const float* __restrict__ zone_w,
                            const float* __restrict__ ig_w, const float* __restrict__ ig_b,
                            float* __restrict__ jz){
    int n = blockIdx.x, bs = blockIdx.y;
    int s = bs & 255;
    int tid = threadIdx.x;
    __shared__ float red[128];
    __shared__ float ohs[96];
    const float* oh = bagua + (size_t)bs*DIMD + n*HDD;
    float contrib = 0.f;
    if (tid < 96) { float v = oh[tid]; ohs[tid] = v; contrib = v * ig_w[tid]; }
    red[tid] = contrib; __syncthreads();
    for (int o = 64; o > 0; o >>= 1){ if (tid < o) red[tid] += red[tid+o]; __syncthreads(); }
    float g = sigmf(red[0] + ig_b[0]);
    float pc0 = ((float)s / 255.0f)*0.5f + ((float)n / 7.0f)*0.5f;
    float pc1 = (float)(s & 31) / 32.0f;
    if (tid < 96) {
        const float* zw = zone_w + (size_t)(n*98)*96 + tid;
        float he = 0.f;
        for (int k = 0; k < 96; k++) he += ohs[k]*zw[(size_t)k*96];
        he += pc0*zw[(size_t)96*96] + pc1*zw[(size_t)97*96];
        float ov = ohs[tid];
        jz[(size_t)bs*DIMD + n*HDD + tid] = ov*(1.f-g) + he*g;
    }
}

__global__ void scores_kernel(const float* __restrict__ jz, const float* __restrict__ xin,
                              const float* __restrict__ w1, const float* __restrict__ b1,
                              const float* __restrict__ w2, const float* __restrict__ b2,
                              float* __restrict__ mask){
    int n = blockIdx.x, tid = threadIdx.x;
    __shared__ float rows[128][96];
    __shared__ float red[128];
    __shared__ float s_ld;
    float sld = 0.f, scd = 0.f;
    const float* W1 = w1 + (size_t)n*96*48;
    const float* W2 = w2 + (size_t)n*48;
    for (int bs = tid; bs < RD; bs += 128) {
        const float* r  = jz  + (size_t)bs*DIMD + n*HDD;
        const float* xi = xin + (size_t)bs*DIMD + n*HDD;
        for (int h = 0; h < 96; h++) { float v = r[h]; rows[tid][h] = v; scd += fabsf(v - xi[h]); }
        float acc = b2[n];
        for (int k = 0; k < 48; k++) {
            float a = b1[n*48 + k];
            for (int h = 0; h < 96; h++) a += rows[tid][h]*W1[(size_t)h*48 + k];
            acc += geluf(a)*W2[k];
        }
        sld += sigmf(acc);
    }
    red[tid] = sld; __syncthreads();
    for (int o = 64; o > 0; o >>= 1){ if (tid < o) red[tid] += red[tid+o]; __syncthreads(); }
    if (tid == 0) s_ld = red[0] / 2048.0f;
    __syncthreads();
    red[tid] = scd; __syncthreads();
    for (int o = 64; o > 0; o >>= 1){ if (tid < o) red[tid] += red[tid+o]; __syncthreads(); }
    if (tid == 0) {
        float cd = red[0] / (2048.0f*96.0f);
        float score = s_ld * sigmf(cd*10.0f - 1.0f);
        mask[n] = (score < 0.3f) ? 0.0f : 1.0f;
    }
}

__global__ void mask_kernel(float* __restrict__ jz, const float* __restrict__ mask){
    int idx = blockIdx.x*256 + threadIdx.x;
    if (idx >= RD*DIMD) return;
    int n = (idx % DIMD) / HDD;
    jz[idx] *= mask[n];
}

// dgsm <- jz + dg*(cg*x)   (dgsm holds dg on entry)
__global__ void smoothed_kernel(const float* __restrict__ jz, const float* __restrict__ cg,
                                const float* __restrict__ x, float* __restrict__ dgsm){
    int idx = blockIdx.x*256 + threadIdx.x;
    if (idx >= RD*DIMD) return;
    dgsm[idx] = jz[idx] + dgsm[idx]*(cg[idx]*x[idx]);
}

// elementwise recurrence over t; RM[b,t,k] = rw * m_in
__global__ void scan_kernel(const float* __restrict__ RW, const float* __restrict__ WW,
                            const float* __restrict__ NM, float* __restrict__ RM){
    int tid = threadIdx.x;     // 256 = 8*32
    int b = tid >> 5, k = tid & 31;
    float m = 0.f;
    for (int t = 0; t < 256; t++) {
        size_t idx = (size_t)(b*256 + t)*32 + k;
        RM[idx] = RW[idx] * m;
        float w = WW[idx];
        m = (1.f - w)*m + w*NM[idx];
    }
}

__global__ void concat_kernel(const float* __restrict__ x, const float* __restrict__ mv,
                              float* __restrict__ cat){
    int idx = blockIdx.x*256 + threadIdx.x;   // exact grid RD*1536
    int r = idx / 1536, c = idx % 1536;
    cat[idx] = (c < DIMD) ? x[(size_t)r*DIMD + c] : mv[(size_t)r*DIMD + (c - DIMD)];
}

// ---------------------------------------------------------------------------
// Host side
// ---------------------------------------------------------------------------
static void gemm(hipStream_t st, int act, const float* A, const h16* Bw, float* C,
                 int M, int N, int K, const float* bias, const float* res, const float* cs){
    dim3 g((N + 127)/128, M/128);
    switch (act) {
        case 0: gemm_kernel<0><<<g,256,0,st>>>(A,Bw,C,M,N,K,bias,res,cs); break;
        case 1: gemm_kernel<1><<<g,256,0,st>>>(A,Bw,C,M,N,K,bias,res,cs); break;
        case 2: gemm_kernel<2><<<g,256,0,st>>>(A,Bw,C,M,N,K,bias,res,cs); break;
        default:gemm_kernel<3><<<g,256,0,st>>>(A,Bw,C,M,N,K,bias,res,cs); break;
    }
}

extern "C" void kernel_launch(void* const* d_in, const int* in_sizes, int n_in,
                              void* d_out, int out_size, void* d_ws, size_t ws_size,
                              hipStream_t stream) {
    (void)in_sizes; (void)n_in; (void)out_size; (void)ws_size;
    const int*   ids      = (const int*)  d_in[0];
    const float* emb      = (const float*)d_in[1];
    const float* pos      = (const float*)d_in[2];
    const float* trig_w   = (const float*)d_in[3];
    const float* res_freq = (const float*)d_in[4];
    const float* pol_w    = (const float*)d_in[5];
    const float* pol_b    = (const float*)d_in[6];
    const float* imp_w1   = (const float*)d_in[7];
    const float* imp_b1   = (const float*)d_in[8];
    const float* imp_w2   = (const float*)d_in[9];
    const float* imp_b2   = (const float*)d_in[10];
    const float* out_w    = (const float*)d_in[11];
    const float* out_b    = (const float*)d_in[12];
    const float* ln1_g    = (const float*)d_in[13];
    const float* ln1_b    = (const float*)d_in[14];
    const float* ff_ln_g  = (const float*)d_in[15];
    const float* ff_ln_b  = (const float*)d_in[16];
    const float* ff_w1    = (const float*)d_in[17];
    const float* ff_b1    = (const float*)d_in[18];
    const float* ff_w2    = (const float*)d_in[19];
    const float* ff_b2    = (const float*)d_in[20];
    const float* sh_w1    = (const float*)d_in[21];
    const float* sh_b1    = (const float*)d_in[22];
    const float* sh_w2    = (const float*)d_in[23];
    const float* sh_b2    = (const float*)d_in[24];
    const float* zone_w   = (const float*)d_in[25];
    const float* ig_w     = (const float*)d_in[26];
    const float* ig_b     = (const float*)d_in[27];
    const float* tsa_w1   = (const float*)d_in[28];
    const float* tsa_b1   = (const float*)d_in[29];
    const float* tsa_w2   = (const float*)d_in[30];
    const float* tsa_b2   = (const float*)d_in[31];
    const float* buf_cg_w = (const float*)d_in[32];
    const float* buf_cg_b = (const float*)d_in[33];
    const float* buf_dg_w = (const float*)d_in[34];
    const float* buf_dg_b = (const float*)d_in[35];
    const float* z_rw_w   = (const float*)d_in[36];
    const float* z_rw_b   = (const float*)d_in[37];
    const float* z_ww_w   = (const float*)d_in[38];
    const float* z_ww_b   = (const float*)d_in[39];
    const float* z_mt_w   = (const float*)d_in[40];
    const float* z_mt_b   = (const float*)d_in[41];
    const float* z_mp_w   = (const float*)d_in[42];
    const float* z_mp_b   = (const float*)d_in[43];
    const float* z_og_w   = (const float*)d_in[44];
    const float* z_og_b   = (const float*)d_in[45];
    const float* head_ln_g= (const float*)d_in[46];
    const float* head_ln_b= (const float*)d_in[47];
    const float* head_w   = (const float*)d_in[48];

    // ---- workspace carve ----
    char* w = (char*)d_ws;
    size_t off = 0;
    auto carve = [&](size_t bytes)->char*{
        char* p = w + off;
        off = (off + bytes + 255) & ~(size_t)255;
        return p;
    };
    h16* hTrig = (h16*)carve((size_t)LLAY*DIMD*DIMD*2);
    h16* hOut  = (h16*)carve((size_t)LLAY*DIMD*DIMD*2);
    h16* hFf1  = (h16*)carve((size_t)LLAY*DIMD*4*DIMD*2);
    h16* hFf2  = (h16*)carve((size_t)LLAY*4*DIMD*DIMD*2);
    h16* hCg   = (h16*)carve((size_t)DIMD*DIMD*2);
    h16* hDg   = (h16*)carve((size_t)DIMD*DIMD*2);
    h16* hZrw  = (h16*)carve((size_t)DIMD*32*2);
    h16* hZww  = (h16*)carve((size_t)DIMD*32*2);
    h16* hZmt  = (h16*)carve((size_t)DIMD*32*2);
    h16* hZmp  = (h16*)carve((size_t)32*DIMD*2);
    h16* hZog  = (h16*)carve((size_t)2*DIMD*DIMD*2);
    h16* hHead = (h16*)carve((size_t)DIMD*VOCABD*2);
    float* f_x  = (float*)carve((size_t)RD*DIMD*4);
    float* f_A  = (float*)carve((size_t)RD*DIMD*4);
    float* f_B  = (float*)carve((size_t)RD*DIMD*4);
    float* f_C  = (float*)carve((size_t)RD*DIMD*4);
    float* f_D  = (float*)carve((size_t)RD*DIMD*4);
    float* f_ff = (float*)carve((size_t)RD*4*DIMD*4);   // also used as concat [RD,1536]
    float* f_factor = (float*)carve(256*4);
    float* f_cs     = (float*)carve((size_t)LLAY*DIMD*4);
    float* f_hm     = (float*)carve((size_t)8*DIMD*4);
    float* f_coef   = (float*)carve(512*4);
    float* f_mask   = (float*)carve(8*4);
    float* f_rw     = (float*)carve((size_t)RD*32*4);
    float* f_ww     = (float*)carve((size_t)RD*32*4);
    float* f_nm     = (float*)carve((size_t)RD*32*4);
    float* f_rm     = (float*)carve((size_t)RD*32*4);

    // ---- weight conversions (fp32 -> f16) ----
    #define CVT(src, dst, n) f32_to_f16_kernel<<<4096,256,0,stream>>>((src),(dst),(size_t)(n))
    trig_perm_kernel<<<(LLAY*DIMD*DIMD + 255)/256, 256, 0, stream>>>(trig_w, hTrig);
    CVT(out_w,    hOut,  (size_t)LLAY*DIMD*DIMD);
    CVT(ff_w1,    hFf1,  (size_t)LLAY*DIMD*4*DIMD);
    CVT(ff_w2,    hFf2,  (size_t)LLAY*4*DIMD*DIMD);
    CVT(buf_cg_w, hCg,   (size_t)DIMD*DIMD);
    CVT(buf_dg_w, hDg,   (size_t)DIMD*DIMD);
    CVT(z_rw_w,   hZrw,  (size_t)DIMD*32);
    CVT(z_ww_w,   hZww,  (size_t)DIMD*32);
    CVT(z_mt_w,   hZmt,  (size_t)DIMD*32);
    CVT(z_mp_w,   hZmp,  (size_t)32*DIMD);
    CVT(z_og_w,   hZog,  (size_t)2*DIMD*DIMD);
    CVT(head_w,   hHead, (size_t)DIMD*VOCABD);
    #undef CVT
    colscale_kernel<<<(LLAY*DIMD + 255)/256, 256, 0, stream>>>(res_freq, f_cs);

    // ---- embedding + scene/causal factor ----
    embed_kernel<<<RD*DIMD/256, 256, 0, stream>>>(ids, emb, pos, f_x);
    scene_kernel<<<1, 256, 0, stream>>>(f_x, tsa_w1, tsa_b1, tsa_w2, tsa_b2, f_factor);

    const int EW = RD*DIMD/256;  // elementwise grid

    for (int l = 0; l < LLAY; l++) {
        // heads = (x @ trigW) * cos(res_freq*pi)         -> f_A
        gemm(stream, 0, f_x, hTrig + (size_t)l*DIMD*DIMD, f_A, RD, DIMD, DIMD,
             nullptr, nullptr, f_cs + (size_t)l*DIMD);
        hm_kernel<<<24, 256, 0, stream>>>(f_A, f_hm);
        coef_kernel<<<1, 256, 0, stream>>>(f_hm,
            pol_w + (size_t)l*NHD*HDD*32, pol_b + (size_t)l*NHD*32,
            imp_w1 + (size_t)l*16, imp_b1 + (size_t)l*16,
            imp_w2 + (size_t)l*16, imp_b2 + (size_t)l, f_coef);
        // merged = heads + transfer                       -> f_B
        transfer_kernel<<<EW, 256, 0, stream>>>(f_A, f_coef, f_factor, f_B);
        // merged @ out_w + out_b + residual(x)            -> f_C ; LN -> bagua f_B
        gemm(stream, 0, f_B, hOut + (size_t)l*DIMD*DIMD, f_C, RD, DIMD, DIMD,
             out_b + (size_t)l*DIMD, f_x, nullptr);
        ln_kernel<<<RD, 256, 0, stream>>>(f_C, f_B, ln1_g + (size_t)l*DIMD, ln1_b + (size_t)l*DIMD);
        // zone encode + input gate                        -> jz in f_A
        zone_kernel<<<dim3(NHD, RD), 128, 0, stream>>>(f_B, zone_w, ig_w, ig_b, f_A);
        // head elimination scores -> mask, then apply
        scores_kernel<<<NHD, 128, 0, stream>>>(f_A, f_x,
            sh_w1 + (size_t)l*NHD*HDD*48, sh_b1 + (size_t)l*NHD*48,
            sh_w2 + (size_t)l*NHD*48,     sh_b2 + (size_t)l*NHD, f_mask);
        mask_kernel<<<EW, 256, 0, stream>>>(f_A, f_mask);
        // buffer gates
        gemm(stream, 2, f_x, hCg, f_C, RD, DIMD, DIMD, buf_cg_b, nullptr, nullptr); // cg
        gemm(stream, 2, f_A, hDg, f_D, RD, DIMD, DIMD, buf_dg_b, nullptr, nullptr); // dg
        smoothed_kernel<<<EW, 256, 0, stream>>>(f_A, f_C, f_x, f_D);  // f_D = smoothed
        // FFN (pre-LN residual)
        ln_kernel<<<RD, 256, 0, stream>>>(f_D, f_C, ff_ln_g + (size_t)l*DIMD, ff_ln_b + (size_t)l*DIMD);
        gemm(stream, 1, f_C, hFf1 + (size_t)l*DIMD*4*DIMD, f_ff, RD, 4*DIMD, DIMD,
             ff_b1 + (size_t)l*4*DIMD, nullptr, nullptr);
        gemm(stream, 0, f_ff, hFf2 + (size_t)l*4*DIMD*DIMD, f_x, RD, DIMD, 4*DIMD,
             ff_b2 + (size_t)l*DIMD, f_D, nullptr);
    }

    // ---- gated memory scan (factored into parallel GEMMs + tiny recurrence) ----
    gemm(stream, 2, f_x, hZrw, f_rw, RD, 32, DIMD, z_rw_b, nullptr, nullptr);
    gemm(stream, 2, f_x, hZww, f_ww, RD, 32, DIMD, z_ww_b, nullptr, nullptr);
    gemm(stream, 0, f_x, hZmt, f_nm, RD, 32, DIMD, z_mt_b, nullptr, nullptr);
    scan_kernel<<<1, 256, 0, stream>>>(f_rw, f_ww, f_nm, f_rm);
    gemm(stream, 0, f_rm, hZmp, f_A, RD, DIMD, 32, z_mp_b, nullptr, nullptr);   // mv
    concat_kernel<<<RD*2*DIMD/256, 256, 0, stream>>>(f_x, f_A, f_ff);
    gemm(stream, 3, f_ff, hZog, f_B, RD, DIMD, 2*DIMD, z_og_b, nullptr, nullptr); // fused

    // ---- head ----
    ln_kernel<<<RD, 256, 0, stream>>>(f_B, f_C, head_ln_g, head_ln_b);
    gemm(stream, 0, f_C, hHead, (float*)d_out, RD, VOCABD, DIMD, nullptr, nullptr, nullptr);
}
